// CrossEntropyLoss_Sub_74586402063357
// MI455X (gfx1250) — matrix operand order repaired and checked by reading, not compiled, not run on hardware
//
#include <hip/hip_runtime.h>
#include <hip/hip_fp16.h>

typedef __attribute__((ext_vector_type(16))) _Float16 v16h;
typedef __attribute__((ext_vector_type(8)))  float    v8f;

#define NB   2
#define NC   19
#define HH   512
#define WW   512
#define GD   64          // down grid  64x64   (factor 8)
#define GS   128         // sup  grid 128x128  (factor 4)
#define PD   (GD*GD)     // 4096
#define PS   (GS*GS)     // 16384
#define KPAD 32
#define SIMT 0.98f
#define EPSV 1e-5f
#define CLMP 10.0f
#define MT   2           // s-tiles (16 rows each) per wave in k_count

// per-pixel side info: pre-scaled threshold + nearest-neighbor label
struct alignas(8) PInfo { float t; int lab; };

// ---------------------------------------------------------------------------
// Block-wide sum (blockDim.x multiple of 32, <=512)
// ---------------------------------------------------------------------------
__device__ __forceinline__ float blockSum(float v) {
#pragma unroll
  for (int off = 16; off; off >>= 1) v += __shfl_xor(v, off, 32);
  __shared__ float ws[32];
  int lane = threadIdx.x & 31, w = threadIdx.x >> 5;
  if (lane == 0) ws[w] = v;
  __syncthreads();
  float s = 0.f;
  if (w == 0) {
    int nw = (blockDim.x + 31) >> 5;
    s = (lane < nw) ? ws[lane] : 0.f;
#pragma unroll
    for (int off = 8; off; off >>= 1) s += __shfl_xor(s, off, 32);
  }
  __syncthreads();
  return s;  // valid in wave 0
}

// ---------------------------------------------------------------------------
// Bilinear downsample (align_corners=false, integer factor => 2x2 average at
// fixed offset) + pack K-major f16 rows [P][32] + fused side-info record
//   info = { 0.98*(|v|^2 + eps),  label[fac*y][fac*x] }   (nearest = ::fac)
//   factor 8 -> samples rows/cols {8y+3, 8y+4};  factor 4 -> {4y+1, 4y+2}
// ---------------------------------------------------------------------------
__global__ void k_pack(const float* __restrict__ src, const int* __restrict__ label,
                       _Float16* __restrict__ outA, PInfo* __restrict__ outI,
                       int grid, int fac, int off) {
  int t = blockIdx.x * blockDim.x + threadIdx.x;
  int P = grid * grid;
  if (t >= NB * P) return;
  int n = t / P, p = t % P;
  int y = p / grid, x = p % grid;
  const float* b = src + (((size_t)n * NC) * HH + (fac * y + off)) * WW + (fac * x + off);
  _Float16* row = outA + (size_t)t * KPAD;
  float acc = 0.f;
#pragma unroll
  for (int c = 0; c < NC; ++c) {
    const float* pc = b + (size_t)c * HH * WW;
    float v = 0.25f * (pc[0] + pc[1] + pc[WW] + pc[WW + 1]);
    acc += v * v;
    row[c] = (_Float16)v;
  }
#pragma unroll
  for (int c = NC; c < KPAD; ++c) row[c] = (_Float16)0.f;
  PInfo pi;
  pi.t   = SIMT * (acc + EPSV);
  pi.lab = label[((size_t)n * HH + fac * y) * WW + fac * x];
  outI[t] = pi;
}

// ---------------------------------------------------------------------------
// Fused similarity GEMM (WMMA f32<-f16 16x16x32) + threshold count + rsqrt.
// One wave per (batch, 32-row s-block): 2 A fragments resident, 256 d-tiles.
//   cos_sim>0.98 <=> dot>max(tns,tnd) <=> (dot>tns)&&(dot>tnd)   (exact)
// Each raw compare is balloted separately (lowers to v_cmp -> SGPR mask);
// masks are combined with s_and and counted with s_bcnt on the scalar pipe,
// which co-issues with WMMA/VALU. Low-half counts recovered by subtraction
// after the loop (saves one SALU op per row per tile).
// Output: pwRaw[n][s] = 1/sqrt(count+10)
// ---------------------------------------------------------------------------
__global__ void k_count(const _Float16* __restrict__ Apack,
                        const _Float16* __restrict__ Bpack,
                        const PInfo* __restrict__ sinfo,
                        const PInfo* __restrict__ dinfo,
                        float* __restrict__ pwRaw) {
  int wid   = blockIdx.x;                 // 0 .. NB*(PS/(16*MT))-1
  int n     = wid / (PS / (16 * MT));
  int sblk  = (wid % (PS / (16 * MT))) * (16 * MT);
  int l  = threadIdx.x;                   // 0..31
  int lm = l & 15;
  int hi = l >> 4;                        // half-wave select

  // A fragments: 16-bit A 16x32 layout (lane = M row; VGPR pairs hold K)
  union { v16h v; unsigned u[8]; } A[MT];
  float tnsv[MT][8];
  int   labs[MT][8];
#pragma unroll
  for (int m = 0; m < MT; ++m) {
    int sbase = sblk + 16 * m;
    const unsigned* arow =
        (const unsigned*)(Apack + ((size_t)n * PS + sbase + lm) * KPAD);
#pragma unroll
    for (int r = 0; r < 8; ++r) {
      int k0 = ((r & 4) ? 16 : 0) + (hi ? 8 : 0) + 2 * (r & 3);
      A[m].u[r] = arow[k0 >> 1];
    }
#pragma unroll
    for (int r = 0; r < 8; ++r) {
      PInfo si = sinfo[(size_t)n * PS + sbase + r + 8 * hi];
      tnsv[m][r] = si.t;
      labs[m][r] = si.lab;
    }
  }

  // wave-uniform counters (SGPRs): scAll = rows r + r+8, scHi = rows r+8
  unsigned scAll[MT][8], scHi[MT][8];
#pragma unroll
  for (int m = 0; m < MT; ++m)
#pragma unroll
    for (int r = 0; r < 8; ++r) { scAll[m][r] = 0u; scHi[m][r] = 0u; }

  const unsigned* brow0 =
      (const unsigned*)(Bpack + ((size_t)n * PD + lm) * KPAD) + (hi ? 4 : 0);
  const PInfo* dptr = dinfo + (size_t)n * PD + lm;

  for (int dt = 0; dt < PD / 16; ++dt) {
    const unsigned* brow = brow0 + (size_t)dt * 16 * (KPAD / 2);
    union { v16h v; unsigned u[8]; } B;
#pragma unroll
    for (int r = 0; r < 8; ++r) B.u[r] = brow[(r & 4) ? r + 4 : r];   // -> 2x global_load_b128
    if (dt + 1 < PD / 16)   // hide L2->L0 latency for the next B tile
      __builtin_prefetch(brow + 16 * (KPAD / 2), 0, 3);

    PInfo di = dptr[dt * 16];                       // -> 1x global_load_b64
    float tndl = di.t;
    int   labd = di.lab;

#pragma unroll
    for (int m = 0; m < MT; ++m) {
      v8f acc = {};
      acc = __builtin_amdgcn_wmma_f32_16x16x32_f16(
          /*neg_a=*/false, A[m].v, /*neg_b=*/false, B.v,
          /*c_mod=*/(short)0, acc, /*reuse_a=*/false, /*reuse_b=*/false);
#pragma unroll
      for (int r = 0; r < 8; ++r) {
        // ballot(raw cmp) lowers to v_cmp with SGPR dest; combine on SALU
        unsigned m1 = __builtin_amdgcn_ballot_w32(acc[r] > tnsv[m][r]);
        unsigned m2 = __builtin_amdgcn_ballot_w32(acc[r] > tndl);
        unsigned m3 = __builtin_amdgcn_ballot_w32(labs[m][r] == labd);
        unsigned msk = m1 & m2 & m3;
        scAll[m][r] += (unsigned)__builtin_popcount(msk);
        scHi[m][r]  += (unsigned)__builtin_popcount(msk >> 16);
      }
    }
  }

  // counts are wave-uniform scalars; one lane writes 1/sqrt(count+10)
  if (l == 0) {
#pragma unroll
    for (int m = 0; m < MT; ++m)
#pragma unroll
      for (int r = 0; r < 8; ++r) {
        int row = sblk + 16 * m + r;
        unsigned lo = scAll[m][r] - scHi[m][r];
        pwRaw[(size_t)n * PS + row]     = rsqrtf((float)lo + 10.0f);
        pwRaw[(size_t)n * PS + row + 8] = rsqrtf((float)scHi[m][r] + 10.0f);
      }
  }
}

// ---------------------------------------------------------------------------
// Bilinear upsample 128->512 (align_corners=false) + per-block partial sums
// ---------------------------------------------------------------------------
__global__ void k_up(const float* __restrict__ pwRaw, float* __restrict__ pwUp,
                     float* __restrict__ part) {
  int t = blockIdx.x * blockDim.x + threadIdx.x;  // NB*HH*WW threads
  int n = t / (HH * WW);
  int p = t % (HH * WW);
  int y = p / WW, x = p % WW;
  float sy = y * 0.25f - 0.375f, sx = x * 0.25f - 0.375f;
  int y0 = (int)floorf(sy), x0 = (int)floorf(sx);
  float wy = sy - (float)y0, wx = sx - (float)x0;
  int y0c = min(max(y0, 0), GS - 1), y1c = min(max(y0 + 1, 0), GS - 1);
  int x0c = min(max(x0, 0), GS - 1), x1c = min(max(x0 + 1, 0), GS - 1);
  const float* s = pwRaw + (size_t)n * PS;
  float v = (1.f - wy) * ((1.f - wx) * s[y0c * GS + x0c] + wx * s[y0c * GS + x1c]) +
            wy        * ((1.f - wx) * s[y1c * GS + x0c] + wx * s[y1c * GS + x1c]);
  pwUp[t] = v;
  float bs = blockSum(v);
  if (threadIdx.x == 0) part[blockIdx.x] = bs;
}

// per-batch mean from upsample partials (1024 blocks per batch), fixed order
__global__ void k_rmean(const float* __restrict__ part, float* __restrict__ meanAcc) {
  for (int n = 0; n < NB; ++n) {
    float s = 0.f;
    for (int i = threadIdx.x; i < 1024; i += 256) s += part[n * 1024 + i];
    float tot = blockSum(s);
    if (threadIdx.x == 0) meanAcc[n] = tot / (float)(HH * WW);
    __syncthreads();
  }
}

// ---------------------------------------------------------------------------
// Fused per-pixel CE (log-softmax over C=19) * normalized/clamped weight,
// per-block partial sums (deterministic)
// ---------------------------------------------------------------------------
__global__ void k_final(const float* __restrict__ cls, const int* __restrict__ label,
                        const float* __restrict__ pwUp, const float* __restrict__ meanAcc,
                        float* __restrict__ part) {
  int t = blockIdx.x * blockDim.x + threadIdx.x;
  int n = t / (HH * WW);
  int p = t % (HH * WW);
  const float* px = cls + ((size_t)n * NC) * HH * WW + p;
  float xs[NC];
  float m = -3.402823466e+38f;
#pragma unroll
  for (int c = 0; c < NC; ++c) { xs[c] = px[(size_t)c * HH * WW]; m = fmaxf(m, xs[c]); }
  float se = 0.f;
#pragma unroll
  for (int c = 0; c < NC; ++c) se += expf(xs[c] - m);
  int lab = label[t];
  float xl = xs[0];
#pragma unroll
  for (int c = 1; c < NC; ++c) xl = (c == lab) ? xs[c] : xl;  // constant indices only
  float ce = -(xl - m - logf(se));
  float pw = fminf(pwUp[t] / meanAcc[n], CLMP);
  float bs = blockSum(ce * pw * (1.0f / (float)(NB * HH * WW)));
  if (threadIdx.x == 0) part[blockIdx.x] = bs;
}

__global__ void k_rout(const float* __restrict__ part, float* __restrict__ out) {
  float s = 0.f;
  for (int i = threadIdx.x; i < 2048; i += 256) s += part[i];
  float tot = blockSum(s);
  if (threadIdx.x == 0) out[0] = tot;
}

// ---------------------------------------------------------------------------
extern "C" void kernel_launch(void* const* d_in, const int* in_sizes, int n_in,
                              void* d_out, int out_size, void* d_ws, size_t ws_size,
                              hipStream_t stream) {
  const float* cls   = (const float*)d_in[0];
  const int*   label = (const int*)d_in[1];

  char* w = (char*)d_ws;
  size_t off = 0;
  auto alloc = [&](size_t bytes) { void* p = w + off; off = (off + bytes + 255) & ~(size_t)255; return p; };
  _Float16* Apack  = (_Float16*)alloc((size_t)NB * PS * KPAD * 2);  // 2.0 MB
  _Float16* Bpack  = (_Float16*)alloc((size_t)NB * PD * KPAD * 2);  // 0.5 MB
  PInfo*    sinfo  = (PInfo*)   alloc((size_t)NB * PS * sizeof(PInfo));
  PInfo*    dinfo  = (PInfo*)   alloc((size_t)NB * PD * sizeof(PInfo));
  float*    pwRaw  = (float*)   alloc((size_t)NB * PS * 4);
  float*    pwUp   = (float*)   alloc((size_t)NB * HH * WW * 4);    // 2.0 MB
  float*    meanA  = (float*)   alloc(256);
  float*    partA  = (float*)   alloc(2048 * 4);
  float*    partB  = (float*)   alloc(2048 * 4);

  // 1) downsample + pack operands (sup: factor 4 offset 1; down: factor 8 offset 3)
  k_pack<<<(NB * PS + 255) / 256, 256, 0, stream>>>(cls, label, Apack, sinfo, GS, 4, 1);
  k_pack<<<(NB * PD + 255) / 256, 256, 0, stream>>>(cls, label, Bpack, dinfo, GD, 8, 3);
  // 2) fused WMMA similarity + count + rsqrt (one wave per 32-row s-block)
  k_count<<<NB * (PS / (16 * MT)), 32, 0, stream>>>(Apack, Bpack, sinfo, dinfo, pwRaw);
  // 3) upsample weights + mean
  k_up<<<NB * HH * WW / 256, 256, 0, stream>>>(pwRaw, pwUp, partA);
  k_rmean<<<1, 256, 0, stream>>>(partA, meanA);
  // 4) fused CE * weight + deterministic reduction to scalar
  k_final<<<NB * HH * WW / 256, 256, 0, stream>>>(cls, label, pwUp, meanA, partB);
  k_rout<<<1, 256, 0, stream>>>(partB, (float*)d_out);
}
